// GAT_64922725646627
// MI455X (gfx1250) — matrix operand order repaired and checked by reading, not compiled, not run on hardware
//
#include <hip/hip_runtime.h>

// ---------------------------------------------------------------------------
// GAT forward (2-layer, 8 heads) for MI455X / gfx1250, wave32 + WMMA bf16.
//
// Pipeline:
//   1) cast x -> bf16; pack W_heads -> WcatT bf16 [512][512] (B transposed),
//      pack W_out -> WoutT bf16 [64][512] (cols 40..63 zero)
//   2) GEMM1 (WMMA bf16, double-buffered LDS, fully unrolled K, b128 feeds)
//   3) per-(node,head) attention scalars  s_src / s_dst
//   4) per-(edge,head) wave32 scatter: w = exp(-leakyrelu(s)); atomic num/den
//   5) finalize: elu(num/den) -> hcat bf16
//   6) GEMM2 (WMMA bf16):  h2 = hcat @ Wout_pad
//   7) layer-2 scores, edge scatter (D=40), finalize -> d_out [50000][40] f32
// ---------------------------------------------------------------------------

#define N_NODES 50000
#define N_EDGES 800000
#define NFEAT   512
#define NHID    64
#define NHEADS  8
#define NCLASS  40
#define ALPHA   0.2f
#define EPS_GAT 1e-16f

typedef __attribute__((ext_vector_type(16))) __bf16 v16bf;
typedef __attribute__((ext_vector_type(8)))  float  v8f;

union FragBF16 {
  unsigned int u[8];
  uint4 q[2];
  v16bf v;
};

static __device__ __forceinline__ unsigned short f32_to_bf16(float f) {
  unsigned int u = __float_as_uint(f);
  unsigned int r = (u + 0x7FFFu + ((u >> 16) & 1u)) >> 16;  // RNE
  return (unsigned short)r;
}

// ---------------------------------------------------------------------------
// Conversion / packing kernels
// ---------------------------------------------------------------------------
__global__ void cast_f32_to_bf16_k(const float* __restrict__ in,
                                   unsigned short* __restrict__ out, int n) {
  int i = blockIdx.x * blockDim.x + threadIdx.x;
  if (i < n) out[i] = f32_to_bf16(in[i]);
}

// WcatT[n][k] = W_heads[h][k][d],  n = h*64 + d   (B stored transposed)
__global__ void pack_WcatT_k(const float* __restrict__ W_heads,
                             unsigned short* __restrict__ WcatT) {
  int i = blockIdx.x * blockDim.x + threadIdx.x;
  if (i >= NFEAT * (NHEADS * NHID)) return;
  int n = i >> 9;            // output column 0..511
  int k = i & 511;           // K index
  int h = n >> 6, d = n & 63;
  float v = W_heads[((size_t)h * NFEAT + k) * NHID + d];
  WcatT[(size_t)n * NFEAT + k] = f32_to_bf16(v);
}

// WoutT[n][k] = W_out[k][n] for n<40, else 0   (padded to 64 columns)
__global__ void pack_WoutT_k(const float* __restrict__ W_out,
                             unsigned short* __restrict__ WoutT) {
  int i = blockIdx.x * blockDim.x + threadIdx.x;
  if (i >= 64 * NFEAT) return;
  int n = i >> 9;
  int k = i & 511;
  float v = (n < NCLASS) ? W_out[(size_t)k * NCLASS + n] : 0.0f;
  WoutT[(size_t)n * NFEAT + k] = f32_to_bf16(v);
}

// ---------------------------------------------------------------------------
// WMMA bf16 GEMM:  C[M][ldc] = A[M][512](bf16) * B[512][Ncols](bf16)
// B supplied transposed: BT[Ncols][512].  Tile 128x64, K-step 32, K fixed
// at compile time -> fully unrolled steady state: global loads get literal
// instruction offsets (kt*64B), LDS offsets are immediates, no per-step
// address math.  256 threads = 8 waves; each wave computes a 16x64 strip
// (4 WMMA frags).  Double-buffered LDS, register-staged prefetch,
// ds_load_b128 fragment feeds (row pitch 20 dwords: 16B alignment +
// conflict-free bank rotation).
// ---------------------------------------------------------------------------
#define KDIM   512
#define TILE_M 128
#define TILE_N 64
#define TILE_K 32
#define NK     (KDIM / TILE_K)   // 16
#define LDA32  (KDIM / 2)        // dwords per A/BT row
#define LDS_PAD 20   // 16 data dwords + 4 pad: 16B-aligned rows, bank rotation

__global__ __launch_bounds__(256)
void gemm_bf16_wmma_k(const unsigned short* __restrict__ A,
                      const unsigned short* __restrict__ BT,
                      float* __restrict__ C,
                      int M, int ldc) {
  __shared__ alignas(16) unsigned int As[2][TILE_M][LDS_PAD];
  __shared__ alignas(16) unsigned int Bs[2][TILE_N][LDS_PAD];

  const int tid  = threadIdx.x;
  const int wave = tid >> 5;
  const int lane = tid & 31;
  const int m16  = lane & 15;
  const int half = lane >> 4;

  const int rowBase = blockIdx.x * TILE_M;
  const int colBase = blockIdx.y * TILE_N;

  const unsigned int* A32 = (const unsigned int*)A;   // 2 bf16 per dword
  const unsigned int* B32 = (const unsigned int*)BT;

  // Per-thread staging indices (8 dwords of A tile, 4 dwords of B tile)
  // plus loop-invariant global base pointers: per-step displacement kt*16
  // dwords (= kt*64 bytes) folds into the load instruction offset.
  int aRow[8], aCol[8], bRow[4], bCol[4];
  const unsigned int* aPtr[8];
  const unsigned int* bPtr[4];
#pragma unroll
  for (int i = 0; i < 8; ++i) {
    int idx = tid + i * 256;                          // 0..2047
    aRow[i] = idx >> 4;
    aCol[i] = idx & 15;
    int gr = rowBase + aRow[i];
    gr = gr < M ? gr : M - 1;                         // clamp (no branch)
    aPtr[i] = A32 + (size_t)gr * LDA32 + aCol[i];
  }
#pragma unroll
  for (int i = 0; i < 4; ++i) {
    int idx = tid + i * 256;                          // 0..1023
    bRow[i] = idx >> 4;
    bCol[i] = idx & 15;
    bPtr[i] = B32 + (size_t)(colBase + bRow[i]) * LDA32 + bCol[i];
  }

  unsigned int aReg[8], bReg[4];

  auto loadTiles = [&](int kt) {
#pragma unroll
    for (int i = 0; i < 8; ++i) aReg[i] = aPtr[i][kt * 16];
#pragma unroll
    for (int i = 0; i < 4; ++i) bReg[i] = bPtr[i][kt * 16];
  };
  auto storeTiles = [&](int buf) {
#pragma unroll
    for (int i = 0; i < 8; ++i) As[buf][aRow[i]][aCol[i]] = aReg[i];
#pragma unroll
    for (int i = 0; i < 4; ++i) Bs[buf][bRow[i]][bCol[i]] = bReg[i];
  };

  v8f acc[4] = {};

  auto computeTile = [&](int buf) {
    // A fragment (16x32 bf16): fa.u[v] = As[row][(v>>2)*8 + half*4 + (v&3)]
    //   -> two contiguous 4-dword runs at dword offsets half*4 and 8+half*4
    FragBF16 fa;
    {
      const unsigned int* ar = &As[buf][(wave << 4) + m16][0];
      fa.q[0] = *(const uint4*)(ar + (half << 2));
      fa.q[1] = *(const uint4*)(ar + 8 + (half << 2));
    }
    // B fragments (32x16 bf16): fb.u[v] = Bs[n][(v>>2)*8 + half*4 + (v&3)]
    //   -> two contiguous 4-dword runs at dword offsets half*4 and 8+half*4
    FragBF16 fb[4];
#pragma unroll
    for (int t = 0; t < 4; ++t) {
      const unsigned int* br = &Bs[buf][(t << 4) + m16][0];
      fb[t].q[0] = *(const uint4*)(br + (half << 2));
      fb[t].q[1] = *(const uint4*)(br + 8 + (half << 2));
    }
#pragma unroll
    for (int t = 0; t < 4; ++t) {
      acc[t] = __builtin_amdgcn_wmma_f32_16x16x32_bf16(
          false, fa.v, false, fb[t].v, (short)0, acc[t], false, false);
    }
  };

  // Prologue: stage tile 0
  loadTiles(0);
  storeTiles(0);
  __syncthreads();

  // Steady state, fully unrolled: prefetch kt+1 while computing kt
#pragma unroll
  for (int kt = 0; kt < NK - 1; ++kt) {
    const int cur = kt & 1, nxt = cur ^ 1;
    loadTiles(kt + 1);          // global loads in flight during compute
    computeTile(cur);
    storeTiles(nxt);            // writes other buffer: no hazard with reads
    __syncthreads();
  }
  computeTile((NK - 1) & 1);

  // C/D layout: VGPR v -> row (half*8 + v), col = m16
#pragma unroll
  for (int t = 0; t < 4; ++t) {
#pragma unroll
    for (int v = 0; v < 8; ++v) {
      int r = rowBase + (wave << 4) + (half << 3) + v;
      int c = colBase + (t << 4) + m16;
      if (r < M) C[(size_t)r * ldc + c] = acc[t][v];
    }
  }
}

// ---------------------------------------------------------------------------
// Attention scalar kernels
// ---------------------------------------------------------------------------
__global__ void node_scores_l1_k(const float* __restrict__ h_all,
                                 const float* __restrict__ a_heads,
                                 float* __restrict__ ssrc,
                                 float* __restrict__ sdst) {
  int i = blockIdx.x * blockDim.x + threadIdx.x;
  if (i >= N_NODES * NHEADS) return;
  int node = i >> 3, h = i & 7;
  const float* hp = h_all + (size_t)node * NFEAT + h * NHID;
  const float* as = a_heads + (size_t)h * (2 * NHID);
  const float* ad = as + NHID;
  float s0 = 0.f, s1 = 0.f;
#pragma unroll
  for (int d = 0; d < NHID; d += 4) {
    float4 hv = *(const float4*)(hp + d);
    float4 av = *(const float4*)(as + d);
    float4 bv = *(const float4*)(ad + d);
    s0 += hv.x * av.x + hv.y * av.y + hv.z * av.z + hv.w * av.w;
    s1 += hv.x * bv.x + hv.y * bv.y + hv.z * bv.z + hv.w * bv.w;
  }
  ssrc[(size_t)h * N_NODES + node] = s0;
  sdst[(size_t)h * N_NODES + node] = s1;
}

__global__ void node_scores_l2_k(const float* __restrict__ h2,
                                 const float* __restrict__ a_out,
                                 float* __restrict__ ssrc,
                                 float* __restrict__ sdst) {
  int i = blockIdx.x * blockDim.x + threadIdx.x;
  if (i >= N_NODES) return;
  const float* hp = h2 + (size_t)i * 64;  // padded row stride 64, valid 0..39
  float s0 = 0.f, s1 = 0.f;
#pragma unroll
  for (int d = 0; d < NCLASS; ++d) {
    float hv = hp[d];
    s0 += hv * a_out[d];
    s1 += hv * a_out[NCLASS + d];
  }
  ssrc[i] = s0;
  sdst[i] = s1;
}

// ---------------------------------------------------------------------------
// Edge scatter kernels: one wave32 per (edge, head) / per edge.
// Edge index / scores are wave-uniform -> force into SGPRs (scalar loads).
// num/h_all fit in the 192MB L2 -> fp32 atomics resolve in L2.
// ---------------------------------------------------------------------------
__global__ __launch_bounds__(256)
void edge_aggr_l1_k(const int* __restrict__ erow, const int* __restrict__ ecol,
                    const float* __restrict__ ssrc, const float* __restrict__ sdst,
                    const float* __restrict__ h_all,
                    float* __restrict__ num, float* __restrict__ den) {
  long long wid = ((long long)blockIdx.x * blockDim.x + threadIdx.x) >> 5;
  int lane = threadIdx.x & 31;
  if (wid >= (long long)N_EDGES * NHEADS) return;
  int e = __builtin_amdgcn_readfirstlane((int)(wid >> 3));   // wave-uniform
  int h = __builtin_amdgcn_readfirstlane((int)(wid & 7));
  int r = erow[e], c = ecol[e];
  float s  = ssrc[(size_t)h * N_NODES + r] + sdst[(size_t)h * N_NODES + c];
  float lr = s > 0.f ? s : ALPHA * s;
  float w  = expf(-lr);
  if (lane == 0) atomicAdd(&den[(size_t)h * N_NODES + r], w);
  const float* hc = h_all + (size_t)c * NFEAT + h * NHID;
  float*       np = num   + (size_t)r * NFEAT + h * NHID;
  atomicAdd(&np[lane],       w * hc[lane]);
  atomicAdd(&np[lane + 32],  w * hc[lane + 32]);
}

__global__ __launch_bounds__(256)
void edge_aggr_l2_k(const int* __restrict__ erow, const int* __restrict__ ecol,
                    const float* __restrict__ ssrc, const float* __restrict__ sdst,
                    const float* __restrict__ h2,
                    float* __restrict__ num2, float* __restrict__ den2) {
  long long wid = ((long long)blockIdx.x * blockDim.x + threadIdx.x) >> 5;
  int lane = threadIdx.x & 31;
  if (wid >= (long long)N_EDGES) return;
  int e = __builtin_amdgcn_readfirstlane((int)wid);          // wave-uniform
  int r = erow[e], c = ecol[e];
  float s  = ssrc[r] + sdst[c];
  float lr = s > 0.f ? s : ALPHA * s;
  float w  = expf(-lr);
  if (lane == 0) atomicAdd(&den2[r], w);
  const float* hc = h2 + (size_t)c * 64;     // padded stride
  float*       np = num2 + (size_t)r * NCLASS;
  atomicAdd(&np[lane], w * hc[lane]);        // lanes 0..31 cover feats 0..31
  if (lane < NCLASS - 32)
    atomicAdd(&np[32 + lane], w * hc[32 + lane]);
}

// ---------------------------------------------------------------------------
// Finalize kernels
// ---------------------------------------------------------------------------
__global__ void finalize_l1_k(const float* __restrict__ num,
                              const float* __restrict__ den,
                              unsigned short* __restrict__ hcat_bf) {
  int i = blockIdx.x * blockDim.x + threadIdx.x;
  if (i >= N_NODES * NFEAT) return;
  int node = i >> 9, f = i & 511;
  int h = f >> 6;
  float v = num[i] / (den[(size_t)h * N_NODES + node] + EPS_GAT);
  v = v > 0.f ? v : expm1f(v);               // ELU
  hcat_bf[i] = f32_to_bf16(v);
}

__global__ void finalize_l2_k(const float* __restrict__ num2,
                              const float* __restrict__ den2,
                              float* __restrict__ out) {
  int i = blockIdx.x * blockDim.x + threadIdx.x;
  if (i >= N_NODES * NCLASS) return;
  int node = i / NCLASS;
  out[i] = num2[i] / (den2[node] + EPS_GAT);
}

// ---------------------------------------------------------------------------
// Host launcher
// ---------------------------------------------------------------------------
extern "C" void kernel_launch(void* const* d_in, const int* in_sizes, int n_in,
                              void* d_out, int out_size, void* d_ws, size_t ws_size,
                              hipStream_t stream) {
  const float* x       = (const float*)d_in[0];  // [50000,512]
  const float* W_heads = (const float*)d_in[1];  // [8,512,64]
  const float* a_heads = (const float*)d_in[2];  // [8,128]
  const float* W_out   = (const float*)d_in[3];  // [512,40]
  const float* a_out   = (const float*)d_in[4];  // [80]
  const int*   edge    = (const int*)d_in[5];    // [2,800000]
  const int*   erow    = edge;
  const int*   ecol    = edge + N_EDGES;
  float*       out     = (float*)d_out;

  // Bump allocator over d_ws (256B aligned)
  char* p = (char*)d_ws;
  auto alloc = [&](size_t bytes) -> void* {
    void* r = (void*)p;
    p += (bytes + 255) & ~(size_t)255;
    return r;
  };

  unsigned short* x_bf   = (unsigned short*)alloc((size_t)N_NODES * NFEAT * 2);
  unsigned short* WcatT  = (unsigned short*)alloc((size_t)NFEAT * NFEAT * 2);
  unsigned short* WoutT  = (unsigned short*)alloc((size_t)64 * NFEAT * 2);
  float* h_all = (float*)alloc((size_t)N_NODES * NFEAT * 4);
  float* ssrc  = (float*)alloc((size_t)NHEADS * N_NODES * 4);
  float* sdst  = (float*)alloc((size_t)NHEADS * N_NODES * 4);
  float* num   = (float*)alloc((size_t)N_NODES * NFEAT * 4);
  float* den   = (float*)alloc((size_t)NHEADS * N_NODES * 4);
  float* h2    = (float*)alloc((size_t)N_NODES * 64 * 4);
  float* s2s   = (float*)alloc((size_t)N_NODES * 4);
  float* s2d   = (float*)alloc((size_t)N_NODES * 4);
  float* num2  = (float*)alloc((size_t)N_NODES * NCLASS * 4);
  float* den2  = (float*)alloc((size_t)N_NODES * 4);
  unsigned short* hcat_bf = x_bf;  // reuse: x_bf dead after GEMM1

  // ---- Stage 0: conversions / packing
  {
    int n = N_NODES * NFEAT;
    cast_f32_to_bf16_k<<<(n + 255) / 256, 256, 0, stream>>>(x, x_bf, n);
  }
  {
    int n = NFEAT * NFEAT;
    pack_WcatT_k<<<(n + 255) / 256, 256, 0, stream>>>(W_heads, WcatT);
  }
  {
    int n = 64 * NFEAT;
    pack_WoutT_k<<<(n + 255) / 256, 256, 0, stream>>>(W_out, WoutT);
  }

  // Zero accumulators (graph-capture-safe memsets)
  hipMemsetAsync(num,  0, (size_t)N_NODES * NFEAT * 4, stream);
  hipMemsetAsync(den,  0, (size_t)NHEADS * N_NODES * 4, stream);
  hipMemsetAsync(num2, 0, (size_t)N_NODES * NCLASS * 4, stream);
  hipMemsetAsync(den2, 0, (size_t)N_NODES * 4, stream);

  const int mBlocks = (N_NODES + TILE_M - 1) / TILE_M;  // 391

  // ---- Stage 1: GEMM1  h_all = x @ Wcat  (all 8 heads fused, N=512)
  gemm_bf16_wmma_k<<<dim3(mBlocks, NFEAT / TILE_N), 256, 0, stream>>>(
      x_bf, WcatT, h_all, N_NODES, NFEAT);

  // ---- Stage 2: per-(node,head) attention scalars
  {
    int n = N_NODES * NHEADS;
    node_scores_l1_k<<<(n + 255) / 256, 256, 0, stream>>>(h_all, a_heads, ssrc, sdst);
  }

  // ---- Stage 3: edge scatter layer 1 (one wave per (edge,head))
  {
    long long waves = (long long)N_EDGES * NHEADS;
    long long blocks = (waves * 32 + 255) / 256;
    edge_aggr_l1_k<<<(unsigned)blocks, 256, 0, stream>>>(
        erow, ecol, ssrc, sdst, h_all, num, den);
  }

  // ---- Stage 4: finalize layer 1 -> bf16 hcat
  {
    int n = N_NODES * NFEAT;
    finalize_l1_k<<<(n + 255) / 256, 256, 0, stream>>>(num, den, hcat_bf);
  }

  // ---- Stage 5: GEMM2  h2 = hcat @ Wout_pad  (N=64, valid cols 0..39)
  gemm_bf16_wmma_k<<<dim3(mBlocks, 1), 256, 0, stream>>>(
      hcat_bf, WoutT, h2, N_NODES, 64);

  // ---- Stage 6: layer-2 scores
  node_scores_l2_k<<<(N_NODES + 255) / 256, 256, 0, stream>>>(h2, a_out, s2s, s2d);

  // ---- Stage 7: edge scatter layer 2 (one wave per edge)
  {
    long long waves = (long long)N_EDGES;
    long long blocks = (waves * 32 + 255) / 256;
    edge_aggr_l2_k<<<(unsigned)blocks, 256, 0, stream>>>(
        erow, ecol, s2s, s2d, h2, num2, den2);
  }

  // ---- Stage 8: finalize -> d_out [50000, 40] f32
  {
    int n = N_NODES * NCLASS;
    finalize_l2_k<<<(n + 255) / 256, 256, 0, stream>>>(num2, den2, out);
  }
}
